// LocalAttentionTransformer_80573586473163
// MI455X (gfx1250) — matrix operand order repaired and checked by reading, not compiled, not run on hardware
//
#include <hip/hip_runtime.h>

// ---------------------------------------------------------------------------
// Types for CDNA5 WMMA (wave32)
// ---------------------------------------------------------------------------
typedef __attribute__((ext_vector_type(16))) __bf16 v16bf;
typedef __attribute__((ext_vector_type(8)))  __bf16 v8bf;
typedef __attribute__((ext_vector_type(8)))  float  v8f;

#define SEQ     4096
#define BATCH   16
#define BS_TOK  (BATCH * SEQ)       // 65536 tokens
#define DMODEL  256
#define NH      8
#define DH      32
#define WIN     128
#define NWIN    (SEQ / WIN)         // 32
#define DFF_    1024
#define IN_DIM_ 64

#define FLG_RELU 1
#define FLG_RES  2
#define FLG_PE   4

__device__ __forceinline__ v8f wmma_bf16(v16bf a, v16bf b, v8f c) {
  // D = A(16x32) * B(32x16) + C, fp32 accumulate
  return __builtin_amdgcn_wmma_f32_16x16x32_bf16(false, a, false, b, (short)0, c,
                                                 false, false);
}

// A-fragment loader: row-major [*,K] source; lane already offset by +16*g.
// elements 0..15 -> K = base..base+15
__device__ __forceinline__ v16bf load_a_frag(const __bf16* p) {
  v8bf lo = *(const v8bf*)(p);
  v8bf hi = *(const v8bf*)(p + 8);
  return __builtin_shufflevector(lo, hi, 0, 1, 2, 3, 4, 5, 6, 7,
                                 8, 9, 10, 11, 12, 13, 14, 15);
}

// ---------------------------------------------------------------------------
// Generic WMMA GEMM: C[M,N] = A[M,K] (bf16) * W[N,K]^T (bf16) + bias
// One wave -> 16 rows x 64 cols (4 accumulators). 8 waves / block.
// flags: FLG_RELU, FLG_RES (out32 += result), FLG_PE (add pe[row%4096, col])
// ---------------------------------------------------------------------------
__global__ __launch_bounds__(256) void k_gemm(
    const __bf16* __restrict__ A, const __bf16* __restrict__ Wt,
    const float* __restrict__ bias, const float* __restrict__ pe,
    float* __restrict__ out32, __bf16* __restrict__ outbf,
    int M, int N, int K, int flags)
{
  const int wave = (int)((blockIdx.x * blockDim.x + threadIdx.x) >> 5);
  const int lane = (int)(threadIdx.x & 31);
  const int nGroups = N >> 6;               // groups of 64 columns
  const int tm = (wave / nGroups) << 4;
  if (tm >= M) return;                      // wave-uniform guard
  const int tn0 = (wave % nGroups) << 6;
  const int g = lane >> 4, r = lane & 15;

  const v8f z = {0.f, 0.f, 0.f, 0.f, 0.f, 0.f, 0.f, 0.f};
  v8f acc[4];
  acc[0] = z; acc[1] = z; acc[2] = z; acc[3] = z;

  const __bf16* arow = A + (size_t)(tm + r) * K + 16 * g;
  const __bf16* brow[4];
#pragma unroll
  for (int t = 0; t < 4; ++t)
    brow[t] = Wt + (size_t)(tn0 + 16 * t + r) * K + 16 * g;

  for (int k0 = 0; k0 < K; k0 += 32) {
    v16bf a = load_a_frag(arow + k0);
#pragma unroll
    for (int t = 0; t < 4; ++t) {
      v16bf b = *(const v16bf*)(brow[t] + k0);
      acc[t] = wmma_bf16(a, b, acc[t]);
    }
  }

#pragma unroll
  for (int t = 0; t < 4; ++t) {
    const int col = tn0 + 16 * t + r;
    const float bn = bias[col];
#pragma unroll
    for (int v = 0; v < 8; ++v) {
      const int row = tm + (g << 3) + v;
      float val = acc[t][v] + bn;
      if (flags & FLG_PE)   val += pe[((row & (SEQ - 1)) << 8) + col];
      if (flags & FLG_RELU) val = fmaxf(val, 0.0f);
      const size_t idx = (size_t)row * N + col;
      if (flags & FLG_RES)  val += out32[idx];
      if (out32) out32[idx] = val;
      if (outbf) outbf[idx] = (__bf16)val;
    }
  }
}

// ---------------------------------------------------------------------------
// Windowed attention: one block per (batch, window, head). 256 thr = 8 waves.
// qkv: bf16 [BS, 768] = [q|k|v]. att: bf16 [BS, 256].
// ---------------------------------------------------------------------------
__global__ __launch_bounds__(256) void k_attn(const __bf16* __restrict__ qkv,
                                              __bf16* __restrict__ att)
{
  __shared__ __attribute__((aligned(64))) float sS[WIN * WIN];  // 64 KB

  const int blk = (int)blockIdx.x;          // B * NWIN * NH = 4096
  const int h   = blk & (NH - 1);
  const int win = (blk >> 3) & (NWIN - 1);
  const int b   = blk >> 8;
  const int tok0 = b * SEQ + win * WIN;

  const int tid  = (int)threadIdx.x;
  const int wv   = tid >> 5;
  const int lane = tid & 31;
  const int g = lane >> 4, r = lane & 15;
  const float scale = 0.17677669529663687f; // 1/sqrt(32)
  const v8f z = {0.f, 0.f, 0.f, 0.f, 0.f, 0.f, 0.f, 0.f};

  // ---- scores = Q K^T * scale  (rows 16*wv..16*wv+15, all 128 cols) ----
  {
    const __bf16* qrow = qkv + (size_t)(tok0 + wv * 16 + r) * 768 + h * DH + 16 * g;
    v16bf a = load_a_frag(qrow);            // Dh = 32 -> single K-step
#pragma unroll
    for (int ct = 0; ct < 8; ++ct) {
      const __bf16* krow =
          qkv + (size_t)(tok0 + ct * 16 + r) * 768 + DMODEL + h * DH + 16 * g;
      v16bf bb = *(const v16bf*)krow;
      v8f acc = wmma_bf16(a, bb, z);
#pragma unroll
      for (int v = 0; v < 8; ++v)
        sS[(wv * 16 + (g << 3) + v) * WIN + ct * 16 + r] = acc[v] * scale;
    }
  }
  __syncthreads();

  // ---- softmax per row; probs written back in place as bf16 ----
  if (tid < WIN) {
    float* p = sS + tid * WIN;
    float mx = -3.4e38f;
    for (int c = 0; c < WIN; ++c) mx = fmaxf(mx, p[c]);
    float sum = 0.f;
    for (int c = 0; c < WIN; ++c) { float e = __expf(p[c] - mx); p[c] = e; sum += e; }
    const float inv = 1.f / sum;
    volatile float* vp = p;
    volatile unsigned short* pb = (volatile unsigned short*)p;
    for (int c = 0; c < WIN; ++c) {
      float e = vp[c];                      // read fp32 word c
      union { __bf16 bh; unsigned short us; } cv;
      cv.bh = (__bf16)(e * inv);
      pb[c] = cv.us;                        // overwrite word c/2 (already read)
    }
  }
  __syncthreads();

  // ---- O = P V : probs bf16 rows (stride 256 bf16) from LDS ----
  const __bf16* sP = (const __bf16*)sS;
#pragma unroll
  for (int nt = 0; nt < 2; ++nt) {
    v8f acc = z;
    for (int k0 = 0; k0 < WIN; k0 += 32) {
      v16bf a = load_a_frag(sP + (size_t)(wv * 16 + r) * 256 + k0 + 16 * g);
      v16bf bb;
#pragma unroll
      for (int i = 0; i < 16; ++i) {
        const int kk = k0 + (g << 4) + i;
        bb[i] = qkv[(size_t)(tok0 + kk) * 768 + 2 * DMODEL + h * DH + nt * 16 + r];
      }
      acc = wmma_bf16(a, bb, acc);
    }
#pragma unroll
    for (int v = 0; v < 8; ++v) {
      const int row = wv * 16 + (g << 3) + v;
      att[(size_t)(tok0 + row) * DMODEL + h * DH + nt * 16 + r] = (__bf16)acc[v];
    }
  }
}

// ---------------------------------------------------------------------------
// LayerNorm over D=256, one wave32 per row; in-place fp32 + bf16 copy out.
// ---------------------------------------------------------------------------
__global__ __launch_bounds__(256) void k_ln(float* __restrict__ h,
                                            __bf16* __restrict__ hb,
                                            const float* __restrict__ gam,
                                            const float* __restrict__ bet)
{
  const int wave = (int)((blockIdx.x * blockDim.x + threadIdx.x) >> 5);
  const int lane = (int)(threadIdx.x & 31);
  if (wave >= BS_TOK) return;
  float* base = h + (size_t)wave * DMODEL;

  float x[8]; float sum = 0.f;
#pragma unroll
  for (int i = 0; i < 8; ++i) { x[i] = base[lane + 32 * i]; sum += x[i]; }
#pragma unroll
  for (int off = 16; off > 0; off >>= 1) sum += __shfl_xor(sum, off, 32);
  const float mu = sum * (1.0f / 256.0f);

  float sq = 0.f;
#pragma unroll
  for (int i = 0; i < 8; ++i) { float d = x[i] - mu; sq += d * d; }
#pragma unroll
  for (int off = 16; off > 0; off >>= 1) sq += __shfl_xor(sq, off, 32);
  const float inv = rsqrtf(sq * (1.0f / 256.0f) + 1e-5f);

#pragma unroll
  for (int i = 0; i < 8; ++i) {
    const int col = lane + 32 * i;
    const float y = (x[i] - mu) * inv * gam[col] + bet[col];
    base[col] = y;
    hb[(size_t)wave * DMODEL + col] = (__bf16)y;
  }
}

// ---------------------------------------------------------------------------
// Small helpers
// ---------------------------------------------------------------------------
__global__ void k_cvt(const float* __restrict__ in, __bf16* __restrict__ out, int n) {
  const int i = (int)(blockIdx.x * 256 + threadIdx.x);
  if (i < n) out[i] = (__bf16)in[i];
}

__global__ void k_pe(float* __restrict__ pe) {
  const int idx = (int)(blockIdx.x * 256 + threadIdx.x);
  if (idx >= SEQ * DMODEL) return;
  const int s = idx >> 8, d = idx & 255;
  const int i2 = d & ~1;                    // 2*(d/2)
  const float div = __expf((float)i2 * (-9.210340371976184f / 256.0f));
  const float ang = (float)s * div;
  pe[idx] = (d & 1) ? cosf(ang) : sinf(ang);
}

__global__ void k_pool(const float* __restrict__ h, float* __restrict__ pooled) {
  const int idx = (int)(blockIdx.x * 256 + threadIdx.x); // 4096
  const int b = idx >> 8, d = idx & 255;
  const float* p = h + (size_t)b * SEQ * DMODEL + d;
  float s = 0.f;
  for (int t = 0; t < SEQ; ++t) s += p[(size_t)t * DMODEL];
  pooled[idx] = s * (1.0f / SEQ);
}

__global__ __launch_bounds__(256) void k_head(
    const float* __restrict__ pooled,
    const float* __restrict__ hw1, const float* __restrict__ hb1,
    const float* __restrict__ hw2, const float* __restrict__ hb2,
    float* __restrict__ out)
{
  __shared__ float sp[16 * 256];
  __shared__ float sh[16 * 128];
  const int tid = (int)threadIdx.x;
  for (int i = tid; i < 16 * 256; i += 256) sp[i] = pooled[i];
  __syncthreads();
  for (int o = tid; o < 16 * 128; o += 256) {
    const int b = o >> 7, j = o & 127;
    float s = hb1[j];
    const float* w = hw1 + (size_t)j * 256;
    const float* pv = sp + b * 256;
    for (int k = 0; k < 256; ++k) s += pv[k] * w[k];
    sh[o] = fmaxf(s, 0.f);
  }
  __syncthreads();
  for (int o = tid; o < 16 * 24; o += 256) {
    const int b = o / 24, j = o % 24;
    float s = hb2[j];
    const float* w = hw2 + (size_t)j * 128;
    const float* hv = sh + b * 128;
    for (int k = 0; k < 128; ++k) s += hv[k] * w[k];
    out[o] = s;
  }
}

// ---------------------------------------------------------------------------
// Host launcher
// ---------------------------------------------------------------------------
extern "C" void kernel_launch(void* const* d_in, const int* in_sizes, int n_in,
                              void* d_out, int out_size, void* d_ws, size_t ws_size,
                              hipStream_t stream) {
  (void)in_sizes; (void)n_in; (void)out_size; (void)ws_size;

  const float* x     = (const float*)d_in[0];
  const float* w_in  = (const float*)d_in[1];
  const float* b_in  = (const float*)d_in[2];
  const float* qkv_w = (const float*)d_in[3];
  const float* qkv_b = (const float*)d_in[4];
  const float* out_w = (const float*)d_in[5];
  const float* out_b = (const float*)d_in[6];
  const float* ff1_w = (const float*)d_in[7];
  const float* ff1_b = (const float*)d_in[8];
  const float* ff2_w = (const float*)d_in[9];
  const float* ff2_b = (const float*)d_in[10];
  const float* ln1_g = (const float*)d_in[11];
  const float* ln1_b = (const float*)d_in[12];
  const float* ln2_g = (const float*)d_in[13];
  const float* ln2_b = (const float*)d_in[14];
  const float* hw1   = (const float*)d_in[15];
  const float* hb1   = (const float*)d_in[16];
  const float* hw2   = (const float*)d_in[17];
  const float* hb2   = (const float*)d_in[18];

  // --- workspace carve-up (256-byte aligned slabs) ---
  char* base = (char*)d_ws;
  size_t off = 0;
  auto carve = [&](size_t bytes) -> char* {
    char* p = base + off;
    off += (bytes + 255) & ~(size_t)255;
    return p;
  };
  __bf16* xbf   = (__bf16*)carve((size_t)BS_TOK * IN_DIM_ * 2);
  float*  h32   = (float*) carve((size_t)BS_TOK * DMODEL * 4);
  __bf16* hbf   = (__bf16*)carve((size_t)BS_TOK * DMODEL * 2);
  __bf16* bigbf = (__bf16*)carve((size_t)BS_TOK * DFF_ * 2);   // qkv (96MB) / ff1-out (128MB)
  __bf16* attbf = (__bf16*)carve((size_t)BS_TOK * DMODEL * 2);
  float*  pe32  = (float*) carve((size_t)SEQ * DMODEL * 4);
  __bf16* winbf = (__bf16*)carve((size_t)DMODEL * IN_DIM_ * 2);
  __bf16* qkvwb = (__bf16*)carve((size_t)3 * 3 * DMODEL * DMODEL * 2);
  __bf16* outwb = (__bf16*)carve((size_t)3 * DMODEL * DMODEL * 2);
  __bf16* ff1wb = (__bf16*)carve((size_t)3 * DFF_ * DMODEL * 2);
  __bf16* ff2wb = (__bf16*)carve((size_t)3 * DMODEL * DFF_ * 2);
  float*  pooled = (float*)carve((size_t)BATCH * DMODEL * 4);

  auto cvt = [&](const float* src, __bf16* dst, int n) {
    k_cvt<<<(n + 255) / 256, 256, 0, stream>>>(src, dst, n);
  };
  // convert inputs / weights to bf16
  cvt(x,     xbf,   BS_TOK * IN_DIM_);
  cvt(w_in,  winbf, DMODEL * IN_DIM_);
  cvt(qkv_w, qkvwb, 3 * 3 * DMODEL * DMODEL);
  cvt(out_w, outwb, 3 * DMODEL * DMODEL);
  cvt(ff1_w, ff1wb, 3 * DFF_ * DMODEL);
  cvt(ff2_w, ff2wb, 3 * DMODEL * DFF_);

  // positional encoding table
  k_pe<<<(SEQ * DMODEL + 255) / 256, 256, 0, stream>>>(pe32);

  auto gemm = [&](const __bf16* A, const __bf16* W, const float* bias,
                  float* o32, __bf16* obf, const float* pe,
                  int M, int N, int K, int flags) {
    const int waves = (M / 16) * (N / 64);
    k_gemm<<<(waves + 7) / 8, 256, 0, stream>>>(A, W, bias, pe, o32, obf,
                                                M, N, K, flags);
  };

  // input projection + positional encoding -> h32 (fp32) + hbf (bf16)
  gemm(xbf, winbf, b_in, h32, hbf, pe32, BS_TOK, DMODEL, IN_DIM_, FLG_PE);

  for (int l = 0; l < 3; ++l) {
    // QKV projection -> bigbf [BS, 768] bf16
    gemm(hbf, qkvwb + (size_t)l * 3 * DMODEL * DMODEL, qkv_b + l * 3 * DMODEL,
         nullptr, bigbf, nullptr, BS_TOK, 3 * DMODEL, DMODEL, 0);
    // windowed attention -> attbf [BS, 256] bf16
    k_attn<<<BATCH * NWIN * NH, 256, 0, stream>>>(bigbf, attbf);
    // output projection, residual add into h32
    gemm(attbf, outwb + (size_t)l * DMODEL * DMODEL, out_b + l * DMODEL,
         h32, nullptr, nullptr, BS_TOK, DMODEL, DMODEL, FLG_RES);
    // layernorm 1 (in-place h32, refresh hbf)
    k_ln<<<BS_TOK / 8, 256, 0, stream>>>(h32, hbf, ln1_g + l * DMODEL, ln1_b + l * DMODEL);
    // FF1 + relu -> bigbf [BS, 1024] bf16
    gemm(hbf, ff1wb + (size_t)l * DFF_ * DMODEL, ff1_b + l * DFF_,
         nullptr, bigbf, nullptr, BS_TOK, DFF_, DMODEL, FLG_RELU);
    // FF2, residual add into h32
    gemm(bigbf, ff2wb + (size_t)l * DMODEL * DFF_, ff2_b + l * DMODEL,
         h32, nullptr, nullptr, BS_TOK, DMODEL, DFF_, FLG_RES);
    // layernorm 2
    k_ln<<<BS_TOK / 8, 256, 0, stream>>>(h32, hbf, ln2_g + l * DMODEL, ln2_b + l * DMODEL);
  }

  // mean pool + head MLP -> d_out [16, 24]
  k_pool<<<BATCH, 256, 0, stream>>>(h32, pooled);
  k_head<<<1, 256, 0, stream>>>(pooled, hw1, hb1, hw2, hb2, (float*)d_out);
}
